// ObserverLayer_26319559590098
// MI455X (gfx1250) — compile-verified, matrix operand order checked by reading, not run on hardware
//
#include <hip/hip_runtime.h>
#include <hip/hip_bf16.h>
#include <math.h>

#define BATCH   8192
#define HIDDEN  768
#define NQ      8
#define NLAYERS 2
#define LN_EPS  1e-5f
#define PI_F    3.14159265358979323846f

typedef __attribute__((ext_vector_type(16))) _Float16 v16h;
typedef __attribute__((ext_vector_type(8)))  float    v8f;

// ---------------------------------------------------------------------------
// Kernel 1: angles = tanh(hidden @ W_in^T + b_in) * pi, via WMMA f16->f32.
// One wave per 16-row batch tile; N = 16 (cols 8..15 are zero padding).
// A (16x32 f16): lane 0-15 -> M=lane, lanes 16-31 -> M=lane-16, K pairs per
// VGPR: v<4 -> K=2v,2v+1 ; v>=4 -> K=16+2(v-4),... ; +8 for upper lane half.
// B (32x16 f16): symmetric layout with lane&15 = N column.
// D (16x16 f32): VGPR g -> M = g + 8*(lane>=16), N = lane&15.
// ---------------------------------------------------------------------------
__global__ __launch_bounds__(256) void angles_wmma_kernel(
    const float* __restrict__ hidden, const float* __restrict__ W_in,
    const float* __restrict__ b_in, float* __restrict__ angles) {
  const int lane = threadIdx.x & 31;
  const int wave = threadIdx.x >> 5;
  const int tile = blockIdx.x * 8 + wave;      // 512 tiles of 16 rows
  const int row0 = tile * 16;
  const int hi   = (lane >> 4) & 1;            // upper lane half -> K offset +8
  const int mn   = lane & 15;                  // A row within tile == B column

  const float* arow = hidden + (size_t)(row0 + mn) * HIDDEN;
  const int   ncol  = (mn < NQ) ? mn : 0;
  const float bmask = (mn < NQ) ? 1.0f : 0.0f; // zero-pad columns 8..15
  const float* bcol = W_in + (size_t)ncol * HIDDEN;

  v8f c = {};
  for (int kc = 0; kc < HIDDEN; kc += 32) {
    v16h a, b;
#pragma unroll
    for (int v = 0; v < 8; ++v) {
      const int kk = kc + (v < 4 ? 2 * v : 8 + 2 * v) + (hi ? 8 : 0);
      const float2 av = *(const float2*)(arow + kk);
      a[2 * v]     = (_Float16)av.x;
      a[2 * v + 1] = (_Float16)av.y;
      const float2 bv = *(const float2*)(bcol + kk);
      b[2 * v]     = (_Float16)(bv.x * bmask);
      b[2 * v + 1] = (_Float16)(bv.y * bmask);
    }
    c = __builtin_amdgcn_wmma_f32_16x16x32_f16(
        /*neg_a=*/false, a, /*neg_b=*/false, b,
        /*c_mod=*/(short)0, c, /*reuse_a=*/false, /*reuse_b=*/false);
  }

  if (mn < NQ) {                               // mn is the qubit column here
    const float bb = b_in[mn];
#pragma unroll
    for (int g = 0; g < 8; ++g) {
      const int m = g + (hi ? 8 : 0);
      angles[(size_t)(row0 + m) * NQ + mn] = tanhf(c[g] + bb) * PI_F;
    }
  }
}

// ---------------------------------------------------------------------------
// Kernel 2: 8-qubit statevector sim (one wave / sample, state in VGPRs) +
// rank-8 output GEMM + LayerNorm, fully fused.
// Amplitude index idx[7:0]: idx[7:5] = register r (8 per lane), idx[4:0]=lane.
// Wire w (0 = MSB) lives at bit B = 7 - w.
// ---------------------------------------------------------------------------
template <int B>
__device__ __forceinline__ void ry_q(float (&re)[8], float (&im)[8],
                                     float c, float s, int lane) {
  if constexpr (B >= 5) {                      // pair across registers
    constexpr int o = 1 << (B - 5);
#pragma unroll
    for (int r = 0; r < 8; ++r) {
      if ((r & o) == 0) {
        const int r1 = r | o;
        float a0 = re[r], a1 = re[r1];
        re[r]  = c * a0 - s * a1;
        re[r1] = s * a0 + c * a1;
        a0 = im[r]; a1 = im[r1];
        im[r]  = c * a0 - s * a1;
        im[r1] = s * a0 + c * a1;
      }
    }
  } else {                                     // pair across lanes
    constexpr int m = 1 << B;
    const bool one = (lane & m) != 0;
#pragma unroll
    for (int r = 0; r < 8; ++r) {
      const float pr = __shfl_xor(re[r], m, 32);
      const float pi = __shfl_xor(im[r], m, 32);
      re[r] = one ? (s * pr + c * re[r]) : (c * re[r] - s * pr);
      im[r] = one ? (s * pi + c * im[r]) : (c * im[r] - s * pi);
    }
  }
}

template <int B>
__device__ __forceinline__ void rz_q(float (&re)[8], float (&im)[8],
                                     float c, float s, int lane) {
#pragma unroll
  for (int r = 0; r < 8; ++r) {
    bool one;
    if constexpr (B >= 5) one = ((r >> (B - 5)) & 1) != 0;
    else                  one = ((lane >> B) & 1) != 0;
    const float t  = one ? -s : s;             // bit0: e^{-i t/2}, bit1: e^{+i t/2}
    const float nr = c * re[r] + t * im[r];
    const float ni = c * im[r] - t * re[r];
    re[r] = nr; im[r] = ni;
  }
}

// CNOT(control = wire0 (bit7 => r>=4), target bit BT): flip target where r>=4
template <int BT>
__device__ __forceinline__ void cnot_c0(float (&re)[8], float (&im)[8], int lane) {
  if constexpr (BT >= 5) {
    constexpr int o = 1 << (BT - 5);
#pragma unroll
    for (int r = 4; r < 8; ++r) {
      if ((r & o) == 0) {
        const int r1 = r | o;
        float t = re[r]; re[r] = re[r1]; re[r1] = t;
        t = im[r]; im[r] = im[r1]; im[r1] = t;
      }
    }
  } else {
    constexpr int m = 1 << BT;
#pragma unroll
    for (int r = 4; r < 8; ++r) {
      re[r] = __shfl_xor(re[r], m, 32);
      im[r] = __shfl_xor(im[r], m, 32);
    }
  }
}

// CNOT(control bit BC, target = wire0 (bit7)): swap r <-> r+4 where control=1
template <int BC>
__device__ __forceinline__ void cnot_t0(float (&re)[8], float (&im)[8], int lane) {
  if constexpr (BC >= 5) {
    constexpr int o = 1 << (BC - 5);
#pragma unroll
    for (int r = 0; r < 4; ++r) {
      if ((r & o) != 0) {
        float t = re[r]; re[r] = re[r + 4]; re[r + 4] = t;
        t = im[r]; im[r] = im[r + 4]; im[r + 4] = t;
      }
    }
  } else {
    constexpr int m = 1 << BC;
    const bool one = (lane & m) != 0;
#pragma unroll
    for (int r = 0; r < 4; ++r) {
      float t = re[r];
      re[r]     = one ? re[r + 4] : re[r];
      re[r + 4] = one ? t         : re[r + 4];
      t = im[r];
      im[r]     = one ? im[r + 4] : im[r];
      im[r + 4] = one ? t         : im[r + 4];
    }
  }
}

__device__ __forceinline__ void apply_ry(float (&re)[8], float (&im)[8],
                                         int w, float th, int lane) {
  float s, c; __sincosf(0.5f * th, &s, &c);
  switch (w) {
    case 0: ry_q<7>(re, im, c, s, lane); break;
    case 1: ry_q<6>(re, im, c, s, lane); break;
    case 2: ry_q<5>(re, im, c, s, lane); break;
    case 3: ry_q<4>(re, im, c, s, lane); break;
    case 4: ry_q<3>(re, im, c, s, lane); break;
    case 5: ry_q<2>(re, im, c, s, lane); break;
    case 6: ry_q<1>(re, im, c, s, lane); break;
    default: ry_q<0>(re, im, c, s, lane); break;
  }
}

__device__ __forceinline__ void apply_rz(float (&re)[8], float (&im)[8],
                                         int w, float th, int lane) {
  float s, c; __sincosf(0.5f * th, &s, &c);
  switch (w) {
    case 0: rz_q<7>(re, im, c, s, lane); break;
    case 1: rz_q<6>(re, im, c, s, lane); break;
    case 2: rz_q<5>(re, im, c, s, lane); break;
    case 3: rz_q<4>(re, im, c, s, lane); break;
    case 4: rz_q<3>(re, im, c, s, lane); break;
    case 5: rz_q<2>(re, im, c, s, lane); break;
    case 6: rz_q<1>(re, im, c, s, lane); break;
    default: rz_q<0>(re, im, c, s, lane); break;
  }
}

__global__ __launch_bounds__(256) void qcircuit_out_ln_kernel(
    const float* __restrict__ angles,
    const float* __restrict__ wry, const float* __restrict__ wrz,
    const float* __restrict__ W_out, const float* __restrict__ b_out,
    const float* __restrict__ gamma, const float* __restrict__ beta,
    float* __restrict__ out) {
  const int lane = threadIdx.x & 31;
  const int s    = blockIdx.x * 8 + (threadIdx.x >> 5);   // sample per wave

  float re[8], im[8];
#pragma unroll
  for (int r = 0; r < 8; ++r) { re[r] = 0.0f; im[r] = 0.0f; }
  re[0] = (lane == 0) ? 1.0f : 0.0f;                      // |0...0>

#pragma unroll
  for (int q = 0; q < NQ; ++q)                            // data encoding
    apply_ry(re, im, q, angles[(size_t)s * NQ + q], lane);

  cnot_c0<6>(re, im, lane); cnot_c0<5>(re, im, lane);     // CNOT(0, t), t=1..7
  cnot_c0<4>(re, im, lane); cnot_c0<3>(re, im, lane);
  cnot_c0<2>(re, im, lane); cnot_c0<1>(re, im, lane);
  cnot_c0<0>(re, im, lane);

#pragma unroll
  for (int l = 0; l < NLAYERS; ++l)
#pragma unroll
    for (int q = 0; q < NQ; ++q) {
      apply_ry(re, im, q, wry[l * NQ + q], lane);
      apply_rz(re, im, q, wrz[l * NQ + q], lane);
    }

  cnot_t0<6>(re, im, lane); cnot_t0<5>(re, im, lane);     // CNOT(c, 0), c=1..7
  cnot_t0<4>(re, im, lane); cnot_t0<3>(re, im, lane);
  cnot_t0<2>(re, im, lane); cnot_t0<1>(re, im, lane);
  cnot_t0<0>(re, im, lane);

  // PauliZ expectations
  float p[8];
#pragma unroll
  for (int r = 0; r < 8; ++r) p[r] = re[r] * re[r] + im[r] * im[r];

  float m[8];
#pragma unroll
  for (int w = 0; w < NQ; ++w) {
    const int b = 7 - w;
    float acc = 0.0f;
#pragma unroll
    for (int r = 0; r < 8; ++r) {
      float sgn;
      if (b >= 5) sgn = ((r >> (b - 5)) & 1) ? -1.0f : 1.0f;
      else        sgn = ((lane >> b) & 1)    ? -1.0f : 1.0f;
      acc += sgn * p[r];
    }
#pragma unroll
    for (int off = 16; off >= 1; off >>= 1) acc += __shfl_xor(acc, off, 32);
    m[w] = acc;
  }

  // y = meas @ W_out^T + b_out, fused LayerNorm (24 cols per lane)
  float yv[24], sum = 0.0f, sq = 0.0f;
#pragma unroll
  for (int k = 0; k < 24; ++k) {
    const int j = lane + 32 * k;
    const float* wrow = W_out + (size_t)j * NQ;
    float y = b_out[j];
#pragma unroll
    for (int q = 0; q < NQ; ++q) y += m[q] * wrow[q];
    yv[k] = y; sum += y; sq += y * y;
  }
#pragma unroll
  for (int off = 16; off >= 1; off >>= 1) {
    sum += __shfl_xor(sum, off, 32);
    sq  += __shfl_xor(sq,  off, 32);
  }
  const float mean = sum * (1.0f / HIDDEN);
  const float var  = sq * (1.0f / HIDDEN) - mean * mean;
  const float rinv = rsqrtf(var + LN_EPS);
#pragma unroll
  for (int k = 0; k < 24; ++k) {
    const int j = lane + 32 * k;
    out[(size_t)s * HIDDEN + j] = (yv[k] - mean) * rinv * gamma[j] + beta[j];
  }
}

extern "C" void kernel_launch(void* const* d_in, const int* in_sizes, int n_in,
                              void* d_out, int out_size, void* d_ws, size_t ws_size,
                              hipStream_t stream) {
  (void)in_sizes; (void)n_in; (void)out_size; (void)ws_size;
  const float* hidden = (const float*)d_in[0];
  const float* W_in   = (const float*)d_in[1];
  const float* b_in   = (const float*)d_in[2];
  const float* W_out  = (const float*)d_in[3];
  const float* b_out  = (const float*)d_in[4];
  const float* wry    = (const float*)d_in[5];
  const float* wrz    = (const float*)d_in[6];
  const float* gamma  = (const float*)d_in[7];
  const float* beta   = (const float*)d_in[8];
  float* out    = (float*)d_out;
  float* angles = (float*)d_ws;                 // BATCH*NQ floats = 256 KB

  // 512 row-tiles, 8 waves per block -> 64 blocks
  angles_wmma_kernel<<<BATCH / 16 / 8, 256, 0, stream>>>(hidden, W_in, b_in, angles);
  // one wave per sample, 8 waves per block -> 1024 blocks
  qcircuit_out_ln_kernel<<<BATCH / 8, 256, 0, stream>>>(
      angles, wry, wrz, W_out, b_out, gamma, beta, out);
}
